// Scalar3DHMM_4440996184887
// MI455X (gfx1250) — matrix operand order repaired and checked by reading, not compiled, not run on hardware
//
#include <hip/hip_runtime.h>

// Problem constants (match reference)
#define XYD 32
#define ZD  16
#define VV  2048
#define BB  32
#define TT  64
#define LL  16
#define SS  (XYD*XYD*ZD)   // 16384 states
#define MM  (BB*TT)        // 2048 sentences
#define NEGF -1e30f

typedef _Float16 half8  __attribute__((ext_vector_type(8)));
typedef _Float16 half16 __attribute__((ext_vector_type(16)));
typedef float    float8 __attribute__((ext_vector_type(8)));

// gfx1250 async global->LDS copies (ASYNCcnt-tracked), guarded so the file
// compiles on toolchains without the builtins (and in the host pass).
#ifndef USE_ASYNC
#  if defined(__has_builtin)
#    if __has_builtin(__builtin_amdgcn_global_load_async_to_lds_b128) && \
        __has_builtin(__builtin_amdgcn_s_wait_asynccnt)
#      define USE_ASYNC 1
#    endif
#  endif
#endif
#ifndef USE_ASYNC
#  define USE_ASYNC 0
#endif

// exact parameter types per hipcc diagnostic: vector_size(16) int, AS1/AS3
typedef int i32x4 __attribute__((vector_size(16)));
typedef __attribute__((address_space(1))) i32x4 g_b128;  // global 16-B chunk
typedef __attribute__((address_space(3))) i32x4 l_b128;  // LDS 16-B chunk

// ---------------------------------------------------------------------------
// Kernel 1: per-state logsumexp over vocab; optionally also emit f16 copy of
// the emission row (fused: this kernel already streams all 128 MB once).
// ---------------------------------------------------------------------------
__global__ __launch_bounds__(256) void k_row_lse(const float* __restrict__ w,
                                                 float* __restrict__ rowZ,
                                                 _Float16* __restrict__ w16) {
  __shared__ float red[256];
  const int s = blockIdx.x;
  const int tid = threadIdx.x;
  const float* row = w + (size_t)s * VV;
  float m = NEGF;
  for (int i = tid; i < VV; i += 256) {
    const float v = row[i];
    m = fmaxf(m, v);
    if (w16) w16[(size_t)s * VV + i] = (_Float16)v;
  }
  red[tid] = m; __syncthreads();
  for (int off = 128; off > 0; off >>= 1) {
    if (tid < off) red[tid] = fmaxf(red[tid], red[tid + off]);
    __syncthreads();
  }
  m = red[0]; __syncthreads();
  float sum = 0.f;
  for (int i = tid; i < VV; i += 256) sum += expf(row[i] - m);
  red[tid] = sum; __syncthreads();
  for (int off = 128; off > 0; off >>= 1) {
    if (tid < off) red[tid] += red[tid + off];
    __syncthreads();
  }
  if (tid == 0) rowZ[s] = m + logf(red[0]);
}

// ---------------------------------------------------------------------------
// Kernel 2: log_softmax of transition_w over its 7 offsets, per state.
// ---------------------------------------------------------------------------
__global__ __launch_bounds__(256) void k_logT(const float* __restrict__ tw,
                                              float* __restrict__ logT) {
  const int s = blockIdx.x * 256 + threadIdx.x;
  if (s >= SS) return;
  float v[7], m = NEGF;
  #pragma unroll
  for (int k = 0; k < 7; k++) { v[k] = tw[s * 7 + k]; m = fmaxf(m, v[k]); }
  float sum = 0.f;
  #pragma unroll
  for (int k = 0; k < 7; k++) sum += expf(v[k] - m);
  const float z = m + logf(sum);
  #pragma unroll
  for (int k = 0; k < 7; k++) logT[s * 7 + k] = v[k] - z;
}

// ---------------------------------------------------------------------------
// Kernel 3: logsumexp of prior_w over all S (single block).
// ---------------------------------------------------------------------------
__global__ __launch_bounds__(256) void k_prior_lse(const float* __restrict__ pw,
                                                   float* __restrict__ priorZ) {
  __shared__ float red[256];
  const int tid = threadIdx.x;
  float m = NEGF;
  for (int i = tid; i < SS; i += 256) m = fmaxf(m, pw[i]);
  red[tid] = m; __syncthreads();
  for (int off = 128; off > 0; off >>= 1) {
    if (tid < off) red[tid] = fmaxf(red[tid], red[tid + off]);
    __syncthreads();
  }
  m = red[0]; __syncthreads();
  float sum = 0.f;
  for (int i = tid; i < SS; i += 256) sum += expf(pw[i] - m);
  red[tid] = sum; __syncthreads();
  for (int off = 128; off > 0; off >>= 1) {
    if (tid < off) red[tid] += red[tid + off];
    __syncthreads();
  }
  if (tid == 0) priorZ[0] = m + logf(red[0]);
}

// ---------------------------------------------------------------------------
// Kernel 4: token-count matrix (M x V) in FP16 (counts <= 16, exact in f16).
// ---------------------------------------------------------------------------
__global__ __launch_bounds__(256) void k_counts(const int* __restrict__ stories,
                                                _Float16* __restrict__ cnts) {
  __shared__ int h[VV];
  const int row = blockIdx.x;            // b*T + t
  const int tid = threadIdx.x;
  for (int i = tid; i < VV; i += 256) h[i] = 0;
  __syncthreads();
  if (tid < LL) atomicAdd(&h[stories[row * LL + tid]], 1);
  __syncthreads();
  for (int i = tid; i < VV; i += 256)
    cnts[(size_t)row * VV + i] = (_Float16)h[i];
}

// ---------------------------------------------------------------------------
// Kernel 5: WMMA GEMM  em[m,s] = sum_v counts[m,v]*W[s,v] - 16*rowZ[s]
// Block: 256 threads (8 waves).  Tile: 128 (M) x 64 (N), K in 32-chunks,
// double-buffered LDS stages filled with async global->LDS B128 copies.
// Each wave computes one 16x64 strip: 4 f32 accumulators, shared A fragment.
// W16P=true: B staged from pre-converted f16 emission (pure async copy).
// W16P=false: B staged from f32 emission with in-register convert.
// ---------------------------------------------------------------------------
#define A_STRIDE 40   // halves; 80-B rows keep all ds b128 accesses 16-B aligned
#define B_STRIDE 40

template <bool W16P>
__global__ __launch_bounds__(256) void k_gemm_em(const _Float16* __restrict__ A,   // (M,V) counts f16
                                                 const _Float16* __restrict__ W16, // (S,V) emission f16
                                                 const float*    __restrict__ W32, // (S,V) emission f32
                                                 const float*    __restrict__ rowZ,
                                                 float* __restrict__ em) {         // (M,S)
  __shared__ _Float16 ldsA[2][128 * A_STRIDE];
  __shared__ _Float16 ldsB[2][64 * B_STRIDE];

  const int tid  = threadIdx.x;
  const int wave = tid >> 5;
  const int lane = tid & 31;
  const int mtile = blockIdx.y * 128;
  const int ntile = blockIdx.x * 64;

  float8 c[4];
  #pragma unroll
  for (int j = 0; j < 4; j++)
    #pragma unroll
    for (int i = 0; i < 8; i++) c[j][i] = 0.f;

  const int ar = tid >> 1, ac = (tid & 1) * 16;  // A staging: row, 16-half chunk
  const int bn = tid >> 2, bk = (tid & 3) * 8;   // B staging: state, 8-elem chunk

  const _Float16* gA   = A + (size_t)(mtile + ar) * VV + ac;
  const _Float16* gB16 = W16P ? (W16 + (size_t)(ntile + bn) * VV + bk) : (const _Float16*)0;
  const float*    gB32 = W32 + (size_t)(ntile + bn) * VV + bk;

  auto stage = [&](int buf, int k0) {
#if USE_ASYNC
    {
      g_b128* ga = (g_b128*)(gA + k0);
      l_b128* la = (l_b128*)&ldsA[buf][ar * A_STRIDE + ac];
      __builtin_amdgcn_global_load_async_to_lds_b128(ga, la, 0, 0);
      __builtin_amdgcn_global_load_async_to_lds_b128(ga, la, 16, 0);
      if constexpr (W16P) {
        __builtin_amdgcn_global_load_async_to_lds_b128(
            (g_b128*)(gB16 + k0),
            (l_b128*)&ldsB[buf][bn * B_STRIDE + bk], 0, 0);
      } else {
        const float8 v = *(const float8*)(gB32 + k0);
        half8 h;
        #pragma unroll
        for (int i = 0; i < 8; i++) h[i] = (_Float16)v[i];
        *(half8*)&ldsB[buf][bn * B_STRIDE + bk] = h;
      }
    }
#else
    {
      const half8* g = (const half8*)(gA + k0);
      half8 v0 = g[0], v1 = g[1];
      *(half8*)&ldsA[buf][ar * A_STRIDE + ac]     = v0;
      *(half8*)&ldsA[buf][ar * A_STRIDE + ac + 8] = v1;
      if constexpr (W16P) {
        *(half8*)&ldsB[buf][bn * B_STRIDE + bk] = *(const half8*)(gB16 + k0);
      } else {
        const float8 v = *(const float8*)(gB32 + k0);
        half8 h;
        #pragma unroll
        for (int i = 0; i < 8; i++) h[i] = (_Float16)v[i];
        *(half8*)&ldsB[buf][bn * B_STRIDE + bk] = h;
      }
    }
#endif
  };

  // fragment addressing (ISA 7.12.2 wave32 layouts)
  const int mrow   = wave * 16 + (lane & 15);
  const int kbaseA = (lane >> 4) * 8;    // A: lanes 16-31 hold K+8 / K+24
  const int kbaseB = (lane >> 4) * 8;    // B: lanes 16-31 hold K+8 / K+24

  stage(0, 0);
  int buf = 0;
  for (int k0 = 0; k0 < VV; k0 += 32, buf ^= 1) {
    if (k0 + 32 < VV) {
      stage(buf ^ 1, k0 + 32);       // prefetch next tile into the other buffer
#if USE_ASYNC
      if constexpr (W16P) __builtin_amdgcn_s_wait_asynccnt(3);  // wait current tile only
      else                __builtin_amdgcn_s_wait_asynccnt(2);
#endif
    } else {
#if USE_ASYNC
      __builtin_amdgcn_s_wait_asynccnt(0);
#endif
    }
    __syncthreads();

    // ---- A fragment: two contiguous 16-B ds loads per lane
    const _Float16* ap = &ldsA[buf][mrow * A_STRIDE + kbaseA];
    half8 alo = *(const half8*)ap;
    half8 ahi = *(const half8*)(ap + 16);
    half16 afrag;
    #pragma unroll
    for (int i = 0; i < 8; i++) { afrag[i] = alo[i]; afrag[8 + i] = ahi[i]; }

    // ---- 4 B fragments, 4 WMMAs (A reused)
    #pragma unroll
    for (int j = 0; j < 4; j++) {
      const _Float16* bp = &ldsB[buf][(j * 16 + (lane & 15)) * B_STRIDE + kbaseB];
      half8 blo = *(const half8*)bp;
      half8 bhi = *(const half8*)(bp + 16);
      half16 bfrag;
      #pragma unroll
      for (int i = 0; i < 8; i++) { bfrag[i] = blo[i]; bfrag[8 + i] = bhi[i]; }
      c[j] = __builtin_amdgcn_wmma_f32_16x16x32_f16(
          false, afrag, false, bfrag, (short)0, c[j], false, false);
    }
    __syncthreads();   // everyone done reading buf before it is restaged
  }

  // ---- epilogue: em = dot - L*rowZ[n]  (C layout: VGPR r -> M = r + (lane>=16)*8)
  const int mlo = mtile + wave * 16 + ((lane >> 4) << 3);
  #pragma unroll
  for (int j = 0; j < 4; j++) {
    const int n = ntile + j * 16 + (lane & 15);
    const float z16 = 16.0f * rowZ[n];
    #pragma unroll
    for (int r = 0; r < 8; r++)
      em[(size_t)(mlo + r) * SS + n] = c[j][r] - z16;
  }
}

// ---------------------------------------------------------------------------
// Kernel 6: alpha_0 = em[:,0,:] + log_softmax(prior)
// ---------------------------------------------------------------------------
__global__ __launch_bounds__(256) void k_init(const float* __restrict__ em,
                                              const float* __restrict__ prior,
                                              const float* __restrict__ priorZ,
                                              float* __restrict__ sc) {
  const int idx = blockIdx.x * 256 + threadIdx.x;  // b*S + s
  if (idx >= BB * SS) return;
  const int b = idx / SS, s = idx - b * SS;
  sc[idx] = em[(size_t)(b * TT) * SS + s] + prior[s] - priorZ[0];
}

// ---------------------------------------------------------------------------
// Kernel 7: one forward-scan step (7-neighbor logsumexp stencil).
// ---------------------------------------------------------------------------
__global__ __launch_bounds__(256) void k_step(const float* __restrict__ in,
                                              const float* __restrict__ logT,
                                              const float* __restrict__ em,
                                              int t,
                                              float* __restrict__ out) {
  const int idx = blockIdx.x * 256 + threadIdx.x;
  if (idx >= BB * SS) return;
  const int b = idx / SS, s = idx - b * SS;
  const int z = s >> 10, rem = s & 1023, y = rem >> 5, x = rem & 31;
  const float* ib = in + (size_t)b * SS;

  float v[7];
  v[0] = ib[s] + logT[s * 7 + 0];
  v[1] = (x >= 1)        ? ib[s - 1]    + logT[(s - 1) * 7 + 1]    : NEGF;
  v[2] = (x <= XYD - 2)  ? ib[s + 1]    + logT[(s + 1) * 7 + 2]    : NEGF;
  v[3] = (y >= 1)        ? ib[s - 32]   + logT[(s - 32) * 7 + 3]   : NEGF;
  v[4] = (y <= XYD - 2)  ? ib[s + 32]   + logT[(s + 32) * 7 + 4]   : NEGF;
  v[5] = (z >= 1)        ? ib[s - 1024] + logT[(s - 1024) * 7 + 5] : NEGF;
  v[6] = (z >= 2)        ? ib[s - 2048] + logT[(s - 2048) * 7 + 6] : NEGF;

  float m = v[0];
  #pragma unroll
  for (int k = 1; k < 7; k++) m = fmaxf(m, v[k]);
  float sum = 0.f;
  #pragma unroll
  for (int k = 0; k < 7; k++) sum += expf(v[k] - m);
  out[idx] = m + logf(sum) + em[(size_t)(b * TT + t) * SS + s];
}

// ---------------------------------------------------------------------------
// Kernel 8: final log-likelihood: logsumexp over S per batch.
// ---------------------------------------------------------------------------
__global__ __launch_bounds__(256) void k_final(const float* __restrict__ sc,
                                               float* __restrict__ out) {
  __shared__ float red[256];
  const int b = blockIdx.x, tid = threadIdx.x;
  const float* p = sc + (size_t)b * SS;
  float m = NEGF;
  for (int i = tid; i < SS; i += 256) m = fmaxf(m, p[i]);
  red[tid] = m; __syncthreads();
  for (int off = 128; off > 0; off >>= 1) {
    if (tid < off) red[tid] = fmaxf(red[tid], red[tid + off]);
    __syncthreads();
  }
  m = red[0]; __syncthreads();
  float sum = 0.f;
  for (int i = tid; i < SS; i += 256) sum += expf(p[i] - m);
  red[tid] = sum; __syncthreads();
  for (int off = 128; off > 0; off >>= 1) {
    if (tid < off) red[tid] += red[tid + off];
    __syncthreads();
  }
  if (tid == 0) out[b] = m + logf(red[0]);
}

// ---------------------------------------------------------------------------
extern "C" void kernel_launch(void* const* d_in, const int* in_sizes, int n_in,
                              void* d_out, int out_size, void* d_ws, size_t ws_size,
                              hipStream_t stream) {
  (void)in_sizes; (void)n_in; (void)out_size;
  const int*   stories  = (const int*)d_in[0];
  const float* emission = (const float*)d_in[3];  // (S, V)
  const float* trans    = (const float*)d_in[4];  // (S, 7)
  const float* prior    = (const float*)d_in[5];  // (S,)
  float* out = (float*)d_out;                     // (B,) f32

  // workspace carve-up (all offsets 256-B aligned)
  char* ws = (char*)d_ws;
  size_t off = 0;
  float*    em     = (float*)(ws + off);    off += (size_t)MM * SS * sizeof(float);    // 128 MB
  _Float16* cnts   = (_Float16*)(ws + off); off += (size_t)MM * VV * sizeof(_Float16); //   8 MB
  float*    rowZ   = (float*)(ws + off);    off += (size_t)SS * sizeof(float);
  float*    logT   = (float*)(ws + off);    off += (size_t)SS * 7 * sizeof(float);
  float*    priorZ = (float*)(ws + off);    off += 256;
  float*    scA    = (float*)(ws + off);    off += (size_t)BB * SS * sizeof(float);
  float*    scB    = (float*)(ws + off);    off += (size_t)BB * SS * sizeof(float);
  // optional f16 emission image (64 MB) if the workspace has room
  _Float16* w16 = nullptr;
  if (ws_size >= off + (size_t)SS * VV * sizeof(_Float16))
    w16 = (_Float16*)(ws + off);

  k_row_lse  <<<SS, 256, 0, stream>>>(emission, rowZ, w16);
  k_logT     <<<SS / 256, 256, 0, stream>>>(trans, logT);
  k_prior_lse<<<1, 256, 0, stream>>>(prior, priorZ);
  k_counts   <<<MM, 256, 0, stream>>>(stories, cnts);

  dim3 gg(SS / 64, MM / 128);     // 256 x 16 blocks
  if (w16) k_gemm_em<true> <<<gg, 256, 0, stream>>>(cnts, w16, emission, rowZ, em);
  else     k_gemm_em<false><<<gg, 256, 0, stream>>>(cnts, nullptr, emission, rowZ, em);

  k_init     <<<(BB * SS) / 256, 256, 0, stream>>>(em, prior, priorZ, scA);

  float* cur = scA; float* nxt = scB;
  for (int t = 1; t < TT; t++) {
    k_step   <<<(BB * SS) / 256, 256, 0, stream>>>(cur, logT, em, t, nxt);
    float* tmp = cur; cur = nxt; nxt = tmp;
  }
  k_final    <<<BB, 256, 0, stream>>>(cur, out);
}
